// TF_Patch_43739946943143
// MI455X (gfx1250) — matrix operand order repaired and checked by reading, not compiled, not run on hardware
//
#include <hip/hip_runtime.h>

// ---------------- dims ----------------
#define Bv   2
#define Cv   48
#define Tv   256
#define Qv   65
#define NHv  4
#define Ev   8
#define Dv   12
#define Hv   192
#define G4v  768     // 4*H
#define KCAT 384     // 192 (x) + 192 (h)
#define L1v  62      // Q - 3
#define L2v  253     // T - 3
#define N1v  512     // B*T sequences (intra)
#define N2v  130     // B*Q sequences (inter)
#define FQK  520     // E*Q
#define FQKP 544     // padded to mult of 32
#define FVv  780     // D*Q
#define FVP  784     // padded to mult of 16
#define EPSv 1e-5f

typedef __attribute__((ext_vector_type(16))) __bf16 v16bf;
typedef __attribute__((ext_vector_type(8)))  float  v8f;
typedef __attribute__((ext_vector_type(4)))  int    v4i_t;
typedef __attribute__((address_space(1))) v4i_t* gptr4_t;
typedef __attribute__((address_space(3))) v4i_t* lptr4_t;

#define DEV __device__ __forceinline__

#if defined(__has_builtin)
#if __has_builtin(__builtin_amdgcn_global_load_async_to_lds_b128) && \
    __has_builtin(__builtin_amdgcn_s_wait_asynccnt)
#define LSTM_ASYNC 1
#endif
#endif
#ifndef LSTM_ASYNC
#define LSTM_ASYNC 0
#endif

DEV v8f wmma_bf(v16bf a, v16bf b, v8f c) {
  return __builtin_amdgcn_wmma_f32_16x16x32_bf16(false, a, false, b, (short)0, c,
                                                 false, false);
}

// A-matrix (16x32 bf16) per-lane element -> K index.
// lanes 0-15: K=0..15 ; lanes 16-31: K=16..31 (contiguous)
DEV int a_kidx(int i, int hi) { return (hi << 4) + i; }

// Load A fragment from row-major [16][ld] bf16 memory (LDS or global), cols k0..k0+31.
DEV v16bf load_a_rm(const __bf16* sm, int ld, int k0, int lane) {
  int m = lane & 15, hi = lane >> 4;
  v16bf a;
#pragma unroll
  for (int i = 0; i < 16; ++i) a[i] = sm[m * ld + k0 + a_kidx(i, hi)];
  return a;
}

DEV float sigf(float x) { return 1.f / (1.f + __expf(-x)); }
DEV float tanhfast(float x) {
  x = fminf(fmaxf(x, -15.f), 15.f);
  float e = __expf(2.f * x);
  return (e - 1.f) / (e + 1.f);
}

// =====================================================================
// Weight packing: B-matrix [K][N] bf16 into WMMA-B tile layout.
// Tile (kt,nt) 32x16: lane l holds column n = nt*16+(l&15),
// elements i=0..15 -> k = kt*32 + (l>>4)*16 + i, stored contiguously.
// =====================================================================
__global__ void k_pack_lstm(const float* __restrict__ Wi,
                            const float* __restrict__ Wh,
                            __bf16* __restrict__ out) {
  // K=384 (12 ktiles), N=768 (48 ntiles): 12*48*512 elems
  int idx = blockIdx.x * blockDim.x + threadIdx.x;
  if (idx >= 12 * 48 * 512) return;
  int tile = idx >> 9, lane = (idx >> 4) & 31, i = idx & 15;
  int kt = tile / 48, nt = tile % 48;
  int k = kt * 32 + ((lane >> 4) << 4) + i;
  int n = nt * 16 + (lane & 15);
  float w = (k < Hv) ? Wi[(size_t)n * Hv + k] : Wh[(size_t)n * Hv + (k - Hv)];
  out[idx] = (__bf16)w;
}

__global__ void k_pack_dec(const float* __restrict__ W, __bf16* __restrict__ out) {
  // K=1536 (48 ktiles), N=48 (3 ntiles). feature k = j*384 + d ; B[k][n] = W[d][n][j]
  int idx = blockIdx.x * blockDim.x + threadIdx.x;
  if (idx >= 48 * 3 * 512) return;
  int tile = idx >> 9, lane = (idx >> 4) & 31, i = idx & 15;
  int kt = tile / 3, nt = tile % 3;
  int k = kt * 32 + ((lane >> 4) << 4) + i;
  int n = nt * 16 + (lane & 15);
  int j = k / 384, d = k - j * 384;
  out[idx] = (__bf16)W[((size_t)d * Cv + n) * 4 + j];
}

// =====================================================================
// LN over channels (axes=(1,)) fused with unfold -> bf16 LSTM input
// X[(n*L + l)*192 + c*4 + j] = LN(z)[b,c,pos=l+j]  (pos along Q or T)
// =====================================================================
__global__ __launch_bounds__(256) void k_ln_unfold(const float* __restrict__ z,
                                                   const float* __restrict__ gam,
                                                   const float* __restrict__ bet,
                                                   __bf16* __restrict__ X, int intra) {
  __shared__ float lnv[Cv * Tv];  // [c][pos], pos extent P<=256
  int tid = threadIdx.x, blk = blockIdx.x;
  int b, fix, P, Lx;
  if (intra) { b = blk / Tv; fix = blk % Tv; P = Qv; Lx = L1v; }
  else       { b = blk / Qv; fix = blk % Qv; P = Tv; Lx = L2v; }
  int n = blk;  // sequence id: b*T+t (intra) or b*Q+q (inter)
  for (int p = tid; p < P; p += 256) {
    float s = 0.f, s2 = 0.f;
    for (int c = 0; c < Cv; ++c) {
      size_t zi = intra ? ((((size_t)b * Cv + c) * Tv + fix) * Qv + p)
                        : ((((size_t)b * Cv + c) * Tv + p) * Qv + fix);
      float v = z[zi];
      s += v; s2 += v * v;
    }
    float mu = s / Cv;
    float inv = rsqrtf(s2 / Cv - mu * mu + EPSv);
    for (int c = 0; c < Cv; ++c) {
      size_t zi = intra ? ((((size_t)b * Cv + c) * Tv + fix) * Qv + p)
                        : ((((size_t)b * Cv + c) * Tv + p) * Qv + fix);
      lnv[c * P + p] = (z[zi] - mu) * inv * gam[c] + bet[c];
    }
  }
  __syncthreads();
  int total = Lx * Hv;
  for (int e = tid; e < total; e += 256) {
    int l = e / Hv, f = e - l * Hv;
    int c = f >> 2, j = f & 3;
    X[((size_t)n * Lx + l) * Hv + f] = (__bf16)lnv[c * P + l + j];
  }
}

// =====================================================================
// BiLSTM: 4 waves per workgroup handle 16 sequences, blockIdx.y = direction.
// Per step: gates[16,768] = bias + [x_t | h](16x384 bf16) @ Wcat^T via WMMA,
// 48 n-tiles split 12-per-wave. x_t is double-buffered in LDS and the next
// step's tile is copied with GLOBAL_LOAD_ASYNC_TO_LDS (overlapped with WMMA)
// when the toolchain exposes the builtin; otherwise a synchronous copy.
// h is kept as bf16 in LDS (consumed only by WMMA); c stays f32.
// =====================================================================
__global__ __launch_bounds__(128) void k_lstm(const __bf16* __restrict__ X,
                                              __bf16* __restrict__ Hout,
                                              const __bf16* __restrict__ Wpack2,
                                              const float* __restrict__ bias_f,
                                              const float* __restrict__ bias_b,
                                              int Nseq, int L) {
  int tid = threadIdx.x;
  int lane = tid & 31, wv = tid >> 5;  // wave id 0..3
  int grp = blockIdx.x;
  int dir = blockIdx.y;
  const __bf16* Wpack = Wpack2 + (size_t)dir * (12 * 48 * 512);
  const float* bias = dir ? bias_b : bias_f;
  __shared__ __align__(16) __bf16 sX[2][16 * Hv];
  __shared__ __align__(16) __bf16 sH[16 * Hv];
  __shared__ float shc[16 * Hv];
  __shared__ float sg[16 * G4v];
  const v16bf* wp = (const v16bf*)Wpack;

  auto fetch = [&](int pa, __bf16* sbuf) {
#if LSTM_ASYNC
    // 16 rows x 384 B = 384 chunks of 16 B; per-lane async DMA into LDS.
    for (int ch = tid; ch < 384; ch += 128) {
      int s = ch / 24, o = (ch - s * 24) * 16;
      int seq = grp * 16 + s; if (seq >= Nseq) seq = Nseq - 1;
      const char* g = (const char*)(X + ((size_t)seq * L + pa) * Hv) + o;
      char* d = (char*)sbuf + s * 384 + o;
      __builtin_amdgcn_global_load_async_to_lds_b128(
          (gptr4_t)(v4i_t*)(size_t)g, (lptr4_t)(v4i_t*)d, 0, 0);
    }
#else
    for (int i = tid; i < 16 * Hv; i += 128) {
      int s = i / Hv, f = i - s * Hv;
      int seq = grp * 16 + s; if (seq >= Nseq) seq = Nseq - 1;
      sbuf[i] = X[((size_t)seq * L + pa) * Hv + f];
    }
#endif
  };

  for (int i = tid; i < 16 * Hv; i += 128) { shc[i] = 0.f; sH[i] = (__bf16)0.f; }
  fetch(dir ? L - 1 : 0, &sX[0][0]);
#if LSTM_ASYNC
  __builtin_amdgcn_s_wait_asynccnt(0);
#endif
  __syncthreads();

  for (int l = 0; l < L; ++l) {
    int cur = l & 1;
    int la = dir ? (L - 1 - l) : l;
    if (l + 1 < L) fetch(dir ? (L - 2 - l) : (l + 1), &sX[1 - cur][0]);
    for (int nt = wv * 12; nt < wv * 12 + 12; ++nt) {
      int col = nt * 16 + (lane & 15);
      float bv = bias[col];
      v8f acc = {bv, bv, bv, bv, bv, bv, bv, bv};
#pragma unroll
      for (int kt = 0; kt < 12; ++kt) {
        const __bf16* asrc = (kt < 6) ? &sX[cur][0] : &sH[0];
        int k0 = (kt < 6) ? kt * 32 : (kt - 6) * 32;
        v16bf a = load_a_rm(asrc, Hv, k0, lane);
        v16bf bm = wp[(size_t)(kt * 48 + nt) * 32 + lane];
        acc = wmma_bf(a, bm, acc);
      }
#pragma unroll
      for (int v = 0; v < 8; ++v)
        sg[(v + ((lane >> 4) << 3)) * G4v + col] = acc[v];
    }
    __syncthreads();
    for (int i = tid; i < 16 * Hv; i += 128) {
      int s = i / Hv, hc = i - s * Hv;
      float gi = sg[s * G4v + hc];
      float gf = sg[s * G4v + Hv + hc];
      float gg = sg[s * G4v + 2 * Hv + hc];
      float go = sg[s * G4v + 3 * Hv + hc];
      float cc = sigf(gf) * shc[i] + sigf(gi) * tanhfast(gg);
      float hh = sigf(go) * tanhfast(cc);
      shc[i] = cc;
      __bf16 hb = (__bf16)hh;
      sH[i] = hb;
      int seq = grp * 16 + s;
      if (seq < Nseq)
        Hout[((size_t)seq * L + la) * (2 * Hv) + dir * Hv + hc] = hb;
    }
#if LSTM_ASYNC
    __builtin_amdgcn_s_wait_asynccnt(0);
#endif
    __syncthreads();
  }
}

// =====================================================================
// ConvTranspose1d as masked im2col GEMM [rows=Nseq*OutLen, K=1536] x [1536,48]
// + bias + residual into z. Within a 32-wide k-tile the tap j is constant,
// so each lane's A fragment is a contiguous window of one H row -> gather
// straight from global (L2-resident), no LDS staging, no barriers.
// =====================================================================
__global__ __launch_bounds__(32) void k_deconv(const __bf16* __restrict__ Hm,
                                               const __bf16* __restrict__ Wpack,
                                               const float* __restrict__ bias,
                                               float* __restrict__ z,
                                               int L, int OutLen, int intra) {
  int lane = threadIdx.x;
  int m = lane & 15, hi = lane >> 4;
  int rowbase = blockIdx.x * 16;
  int row = rowbase + m;
  int n = row / OutLen, pos = row - n * OutLen;
  const v16bf* wp = (const v16bf*)Wpack;
  float b0 = bias[m], b1 = bias[16 + m], b2 = bias[32 + m];
  v8f acc0 = {b0, b0, b0, b0, b0, b0, b0, b0};
  v8f acc1 = {b1, b1, b1, b1, b1, b1, b1, b1};
  v8f acc2 = {b2, b2, b2, b2, b2, b2, b2, b2};
  for (int kt = 0; kt < 48; ++kt) {
    int j = kt / 12, dbase = (kt - j * 12) * 32;
    int l = pos - j;
    bool valid = (l >= 0) && (l < L);
    int lc = valid ? l : 0;
    const __bf16* ap = Hm + ((size_t)n * L + lc) * 384 + dbase;
    v16bf a;
#pragma unroll
    for (int i = 0; i < 16; ++i) a[i] = valid ? ap[a_kidx(i, hi)] : (__bf16)0.f;
    acc0 = wmma_bf(a, wp[(size_t)(kt * 3 + 0) * 32 + lane], acc0);
    acc1 = wmma_bf(a, wp[(size_t)(kt * 3 + 1) * 32 + lane], acc1);
    acc2 = wmma_bf(a, wp[(size_t)(kt * 3 + 2) * 32 + lane], acc2);
  }
#define DECONV_EPI(ACC, NT)                                              \
  {                                                                      \
    int c = (NT)*16 + m;                                                 \
    _Pragma("unroll") for (int v = 0; v < 8; ++v) {                      \
      int mo = v + (hi << 3);                                            \
      int ro = rowbase + mo;                                             \
      int no = ro / OutLen, po = ro - no * OutLen;                       \
      int bb, tt, qq;                                                    \
      if (intra) { bb = no / Tv; tt = no % Tv; qq = po; }                \
      else       { bb = no / Qv; qq = no % Qv; tt = po; }                \
      size_t zi = (((size_t)bb * Cv + c) * Tv + tt) * Qv + qq;           \
      z[zi] += ACC[v];                                                   \
    }                                                                    \
  }
  DECONV_EPI(acc0, 0)
  DECONV_EPI(acc1, 1)
  DECONV_EPI(acc2, 2)
#undef DECONV_EPI
}

// =====================================================================
// QKV 1x1 projections + PReLU + LN over (E|D, Q); write padded bf16 rows
// =====================================================================
__global__ __launch_bounds__(256) void k_qkv(
    const float* __restrict__ z,
    const float* WQ, const float* bQ, const float* aQ, const float* gQ, const float* beQ,
    const float* WK, const float* bK, const float* aK, const float* gK, const float* beK,
    const float* WV, const float* bV, const float* aV, const float* gV, const float* beV,
    __bf16* __restrict__ qf, __bf16* __restrict__ kf, __bf16* __restrict__ vf) {
  int tid = threadIdx.x, blk = blockIdx.x;
  int t = blk % Tv, h = (blk / Tv) % NHv, b = blk / (Tv * NHv);
  __shared__ float zb[Cv * Qv];
  __shared__ float buf[Dv * Qv];
  __shared__ float rs[256], rq[256];
  for (int i = tid; i < Cv * Qv; i += 256) {
    int c = i / Qv, qq = i - c * Qv;
    zb[i] = z[(((size_t)b * Cv + c) * Tv + t) * Qv + qq];
  }
  __syncthreads();
  auto do_proj = [&](const float* W, const float* bias, const float* alpha,
                     const float* gamma, const float* beta, int R,
                     __bf16* out, int padF) {
    int M = R * Qv;
    for (int i = tid; i < M; i += 256) {
      int e = i / Qv, qq = i - e * Qv;
      const float* w = W + (size_t)(h * R + e) * Cv;
      float s = bias[h * R + e];
      for (int c = 0; c < Cv; ++c) s += zb[c * Qv + qq] * w[c];
      s = (s >= 0.f) ? s : alpha[h] * s;
      buf[i] = s;
    }
    __syncthreads();
    float ps = 0.f, pq = 0.f;
    for (int i = tid; i < M; i += 256) { float v = buf[i]; ps += v; pq += v * v; }
    rs[tid] = ps; rq[tid] = pq;
    __syncthreads();
    for (int s2 = 128; s2 > 0; s2 >>= 1) {
      if (tid < s2) { rs[tid] += rs[tid + s2]; rq[tid] += rq[tid + s2]; }
      __syncthreads();
    }
    float mu = rs[0] / M;
    float inv = rsqrtf(rq[0] / M - mu * mu + EPSv);
    __syncthreads();
    size_t base = ((size_t)(b * NHv + h) * Tv + t) * padF;
    for (int i = tid; i < padF; i += 256) {
      __bf16 o = (__bf16)0.f;
      if (i < M) {
        int e = i / Qv, qq = i - e * Qv;
        float v = (buf[i] - mu) * inv * gamma[(h * R + e) * Qv + qq] +
                  beta[(h * R + e) * Qv + qq];
        o = (__bf16)v;
      }
      out[base + i] = o;
    }
    __syncthreads();
  };
  do_proj(WQ, bQ, aQ, gQ, beQ, Ev, qf, FQKP);
  do_proj(WK, bK, aK, gK, beK, Ev, kf, FQKP);
  do_proj(WV, bV, aV, gV, beV, Dv, vf, FVP);
}

// ===================== S = qf @ kf^T  (per (b,h)) =====================
__global__ __launch_bounds__(32) void k_sgemm(const __bf16* __restrict__ qf,
                                              const __bf16* __restrict__ kf,
                                              float* __restrict__ S) {
  int lane = threadIdx.x, blk = blockIdx.x;
  int tj = blk & 15, ti = (blk >> 4) & 15, bh = blk >> 8;
  size_t base = (size_t)bh * Tv * FQKP;
  int m = lane & 15, hi = lane >> 4;
  v8f acc = {0.f,0.f,0.f,0.f,0.f,0.f,0.f,0.f};
  for (int kt = 0; kt < FQKP / 32; ++kt) {
    const __bf16* ap = qf + base + (size_t)(ti * 16 + m) * FQKP + kt * 32;
    v16bf a;
#pragma unroll
    for (int i = 0; i < 16; ++i) a[i] = ap[a_kidx(i, hi)];
    const __bf16* bp = kf + base + (size_t)(tj * 16 + m) * FQKP + kt * 32 + (hi << 4);
    v16bf bm;
#pragma unroll
    for (int i = 0; i < 16; ++i) bm[i] = bp[i];
    acc = wmma_bf(a, bm, acc);
  }
  size_t sb = (size_t)bh * Tv * Tv;
  int n = tj * 16 + m;
#pragma unroll
  for (int v = 0; v < 8; ++v)
    S[sb + (size_t)(ti * 16 + v + (hi << 3)) * Tv + n] = acc[v];
}

// ===================== softmax rows of S -> P bf16 =====================
__global__ __launch_bounds__(256) void k_softmax(const float* __restrict__ S,
                                                 __bf16* __restrict__ P) {
  __shared__ float red[256];
  int tid = threadIdx.x;
  size_t idx = (size_t)blockIdx.x * Tv + tid;
  float scale = rsqrtf((float)FQK);
  float v = S[idx] * scale;
  red[tid] = v;
  __syncthreads();
  for (int s = 128; s > 0; s >>= 1) {
    if (tid < s) red[tid] = fmaxf(red[tid], red[tid + s]);
    __syncthreads();
  }
  float mx = red[0];
  __syncthreads();
  float e = __expf(v - mx);
  red[tid] = e;
  __syncthreads();
  for (int s = 128; s > 0; s >>= 1) {
    if (tid < s) red[tid] += red[tid + s];
    __syncthreads();
  }
  P[idx] = (__bf16)(e / red[0]);
}

// ===================== O = P @ vf  (per (b,h)) =====================
__global__ __launch_bounds__(32) void k_ogemm(const __bf16* __restrict__ P,
                                              const __bf16* __restrict__ vf,
                                              float* __restrict__ O) {
  int lane = threadIdx.x, blk = blockIdx.x;
  int nt = blk % 49, ti = (blk / 49) & 15, bh = blk / (49 * 16);
  size_t pb = (size_t)bh * Tv * Tv;
  size_t vb = (size_t)bh * Tv * FVP;
  int m = lane & 15, hi = lane >> 4;
  v8f acc = {0.f,0.f,0.f,0.f,0.f,0.f,0.f,0.f};
  for (int kt = 0; kt < Tv / 32; ++kt) {
    const __bf16* ap = P + pb + (size_t)(ti * 16 + m) * Tv + kt * 32;
    v16bf a;
#pragma unroll
    for (int i = 0; i < 16; ++i) a[i] = ap[a_kidx(i, hi)];
    v16bf bm;
#pragma unroll
    for (int i = 0; i < 16; ++i)
      bm[i] = vf[vb + (size_t)(kt * 32 + (hi << 4) + i) * FVP + nt * 16 + m];
    acc = wmma_bf(a, bm, acc);
  }
  int col = nt * 16 + m;
  if (col < FVv) {
    size_t ob = (size_t)bh * Tv * FVv;
#pragma unroll
    for (int v = 0; v < 8; ++v)
      O[ob + (size_t)(ti * 16 + v + (hi << 3)) * FVv + col] = acc[v];
  }
}

// ====== final: head-merge, 1x1 proj Wp, PReLU, LN over (C,Q), +z ======
__global__ __launch_bounds__(256) void k_final(const float* __restrict__ z,
                                               const float* __restrict__ O,
                                               const float* Wp, const float* bp,
                                               const float* ap, const float* gp,
                                               const float* bep,
                                               float* __restrict__ out) {
  int tid = threadIdx.x, blk = blockIdx.x;
  int b = blk / Tv, t = blk % Tv;
  __shared__ float ov[Cv * Qv], po[Cv * Qv];
  __shared__ float rs[256], rq[256];
  for (int i = tid; i < Cv * Qv; i += 256) {
    int cc = i / Qv, qq = i - cc * Qv;
    int h = cc / Dv, d = cc - h * Dv;
    ov[i] = O[((size_t)(b * NHv + h) * Tv + t) * FVv + d * Qv + qq];
  }
  __syncthreads();
  float a0 = ap[0];
  for (int i = tid; i < Cv * Qv; i += 256) {
    int c = i / Qv, qq = i - c * Qv;
    float s = bp[c];
    for (int cc = 0; cc < Cv; ++cc) s += ov[cc * Qv + qq] * Wp[c * Cv + cc];
    s = (s >= 0.f) ? s : a0 * s;
    po[i] = s;
  }
  __syncthreads();
  float ps = 0.f, pq = 0.f;
  for (int i = tid; i < Cv * Qv; i += 256) { float v = po[i]; ps += v; pq += v * v; }
  rs[tid] = ps; rq[tid] = pq;
  __syncthreads();
  for (int s2 = 128; s2 > 0; s2 >>= 1) {
    if (tid < s2) { rs[tid] += rs[tid + s2]; rq[tid] += rq[tid + s2]; }
    __syncthreads();
  }
  const int M = Cv * Qv;
  float mu = rs[0] / M;
  float inv = rsqrtf(rq[0] / M - mu * mu + EPSv);
  for (int i = tid; i < M; i += 256) {
    int c = i / Qv, qq = i - c * Qv;
    float val = (po[i] - mu) * inv * gp[i] + bep[i];
    size_t zi = (((size_t)b * Cv + c) * Tv + t) * Qv + qq;
    out[zi] = z[zi] + val;
  }
}

// =====================================================================
extern "C" void kernel_launch(void* const* d_in, const int* in_sizes, int n_in,
                              void* d_out, int out_size, void* d_ws, size_t ws_size,
                              hipStream_t stream) {
  (void)in_sizes; (void)n_in; (void)out_size; (void)ws_size;
  const size_t NZ = (size_t)Bv * Cv * Tv * Qv;
  const size_t maxRows = (size_t)N2v * L2v;  // 32890 >= 31744
  char* w = (char*)d_ws;
  size_t off = 0;
  auto alloc = [&](size_t bytes) {
    size_t o = off;
    off += (bytes + 255) & ~(size_t)255;
    return o;
  };
  float*  zb   = (float*)(w + alloc(NZ * sizeof(float)));
  __bf16* Xb   = (__bf16*)(w + alloc(maxRows * Hv * 2));
  __bf16* Hb   = (__bf16*)(w + alloc(maxRows * 2 * Hv * 2));
  __bf16* Wcat = (__bf16*)(w + alloc((size_t)4 * 12 * 48 * 512 * 2));
  __bf16* Wd   = (__bf16*)(w + alloc((size_t)2 * 48 * 3 * 512 * 2));
  __bf16* qf   = (__bf16*)(w + alloc((size_t)Bv * NHv * Tv * FQKP * 2));
  __bf16* kf   = (__bf16*)(w + alloc((size_t)Bv * NHv * Tv * FQKP * 2));
  __bf16* vf   = (__bf16*)(w + alloc((size_t)Bv * NHv * Tv * FVP * 2));
  float*  Sb   = (float*)(w + alloc((size_t)Bv * NHv * Tv * Tv * 4));
  __bf16* Pb   = (__bf16*)(w + alloc((size_t)Bv * NHv * Tv * Tv * 2));
  float*  Ob   = (float*)(w + alloc((size_t)Bv * NHv * Tv * FVv * 4));
  const size_t WCS = (size_t)12 * 48 * 512;  // elems per packed lstm dir
  const size_t WDS = (size_t)48 * 3 * 512;

  for (int s = 0; s < 2; ++s) {
    // jax pytree flatten: params['m'] leaves at d_in[2..41], params['x'] at [42..81]
    int pb = (s == 0) ? 42 : 2;
    auto P = [&](int k) -> const float* { return (const float*)d_in[pb + k]; };
    (void)hipMemcpyAsync(zb, d_in[s], NZ * sizeof(float), hipMemcpyDeviceToDevice,
                         stream);
    // pack weights: lstm_intra f(Wif,Whf)=37,35 b=36,34 ; lstm_inter f=31,29 b=30,28
    k_pack_lstm<<<1152, 256, 0, stream>>>(P(37), P(35), Wcat + 0 * WCS);
    k_pack_lstm<<<1152, 256, 0, stream>>>(P(36), P(34), Wcat + 1 * WCS);
    k_pack_lstm<<<1152, 256, 0, stream>>>(P(31), P(29), Wcat + 2 * WCS);
    k_pack_lstm<<<1152, 256, 0, stream>>>(P(30), P(28), Wcat + 3 * WCS);
    k_pack_dec<<<288, 256, 0, stream>>>(P(22), Wd);           // dec_intra.W
    k_pack_dec<<<288, 256, 0, stream>>>(P(20), Wd + WDS);     // dec_inter.W
    // ---- intra (freq) path ----
    k_ln_unfold<<<Bv * Tv, 256, 0, stream>>>(zb, P(27), P(26), Xb, 1);
    k_lstm<<<dim3((N1v + 15) / 16, 2), 128, 0, stream>>>(Xb, Hb, Wcat + 0 * WCS,
                                                         P(39), P(38), N1v, L1v);
    k_deconv<<<(N1v * Qv) / 16, 32, 0, stream>>>(Hb, Wd, P(23), zb, L1v, Qv, 1);
    // ---- inter (time) path ----
    k_ln_unfold<<<Bv * Qv, 256, 0, stream>>>(zb, P(25), P(24), Xb, 0);
    k_lstm<<<dim3((N2v + 15) / 16, 2), 128, 0, stream>>>(Xb, Hb, Wcat + 2 * WCS,
                                                         P(33), P(32), N2v, L2v);
    k_deconv<<<(N2v * Tv) / 16, 32, 0, stream>>>(Hb, Wd + WDS, P(21), zb, L2v, Tv, 0);
    // ---- attention ----
    k_qkv<<<Bv * NHv * Tv, 256, 0, stream>>>(
        zb, P(1), P(9), P(5), P(17), P(12),   // Q: WQ bQ aQ gQ beQ
        P(0), P(8), P(4), P(16), P(11),       // K
        P(2), P(10), P(6), P(18), P(13),      // V
        qf, kf, vf);
    k_sgemm<<<Bv * NHv * 16 * 16, 32, 0, stream>>>(qf, kf, Sb);
    k_softmax<<<Bv * NHv * Tv, 256, 0, stream>>>(Sb, Pb);
    k_ogemm<<<Bv * NHv * 16 * 49, 32, 0, stream>>>(Pb, vf, Ob);
    k_final<<<Bv * Tv, 256, 0, stream>>>(zb, Ob, P(3), P(15), P(7), P(19), P(14),
                                         (float*)d_out + (size_t)s * NZ);
  }
}
